// Model_66778151518433
// MI455X (gfx1250) — matrix operand order repaired and checked by reading, not compiled, not run on hardware
//
#include <hip/hip_runtime.h>
#include <stdint.h>

// ---------------------------------------------------------------------------
// BiLSTM(2xBiDir,H=384) + FC(T=9) + CRF(NLL + Viterbi) for MI455X (gfx1250).
// Heavy math: bf16 WMMA (v_wmma_f32_16x16x32_bf16), wave32.
// Layer scan = ONE persistent kernel (8 workgroups, each owning 16 batch rows
// of one direction). Pregate tiles are double-buffered into LDS with
// GLOBAL_LOAD_ASYNC_TO_LDS_B128 (ASYNCcnt) to hide the 403MB gate stream
// behind the WMMA recurrence.
// ---------------------------------------------------------------------------

typedef __attribute__((ext_vector_type(16))) __bf16        v16bf;
typedef __attribute__((ext_vector_type(8)))  float         v8f;
typedef __attribute__((ext_vector_type(4)))  unsigned int  u32x4;

#define B_   64
#define S_   512
#define D_   768
#define H_   384
#define G4H  1536     // 4*H
#define T_   9
#define L_   2

// k_scan LDS layout: two unpadded 16x1536 f32 pregate/gate buffers (linear so
// the async B128 copies are address-linear), then padded h staging buffer.
#define PLDS_WORDS  (16 * G4H)                 // 24576 f32 = 96KB per buffer
#define HLDS_STRIDE 392                        // bf16 halves per h row (padded)
#define SMEM_BYTES  (2 * PLDS_WORDS * 4 + 16 * HLDS_STRIDE * 2)   // 209152

__device__ __forceinline__ unsigned short f32_bf16(float f) {
  unsigned int u = __float_as_uint(f);
  u += 0x7FFFu + ((u >> 16) & 1u);             // round-to-nearest-even
  return (unsigned short)(u >> 16);
}
__device__ __forceinline__ float bf16_f32(unsigned short s) {
  return __uint_as_float(((unsigned int)s) << 16);
}

union FragU { v16bf v; u32x4 q[2]; };

// A-matrix 16x32 bf16 fragment (row-major A[m][k]).
// lanes 0-15: M=lane, v0..v3 = K 0..7, v4..v7 = K 16..23; lanes 16-31: K+8.
__device__ __forceinline__ v16bf load_a(const unsigned short* A, int lda,
                                        int m, int k0, int lane) {
  FragU f;
  const unsigned short* p = A + (size_t)m * lda + k0 + ((lane >> 4) << 3);
  f.q[0] = *(const u32x4*)(p);
  f.q[1] = *(const u32x4*)(p + 16);
  return f.v;
}

// B-matrix 32x16 bf16 fragment. B[k][n] = W[n][k], W row-major [N][K].
// lanes 0-15: N=lane, K=k0..k0+7,k0+16..k0+23 ; lanes 16-31: N=lane-16, K+8.
__device__ __forceinline__ v16bf load_b(const unsigned short* W, int ldb,
                                        int n, int k0, int lane) {
  FragU f;
  const unsigned short* p = W + (size_t)n * ldb + k0 + ((lane >> 4) << 3);
  f.q[0] = *(const u32x4*)(p);
  f.q[1] = *(const u32x4*)(p + 16);
  return f.v;
}

#define WMMA_BF16(ACC, A, Bv) \
  (ACC) = __builtin_amdgcn_wmma_f32_16x16x32_bf16(false, (A), false, (Bv), (short)0, (ACC), false, false)

// Async global->LDS copy, tracked with ASYNCcnt (CDNA5 VGLOBAL op 98).
__device__ __forceinline__ void async_ld_b128(unsigned int lds_off, const void* g) {
  asm volatile("global_load_async_to_lds_b128 %0, %1, off"
               :: "v"(lds_off), "v"((unsigned long long)(uintptr_t)g)
               : "memory");
}
__device__ __forceinline__ void wait_async0() {
  asm volatile("s_wait_asynccnt 0x0" ::: "memory");
}

// ---------------------------------------------------------------------------
__global__ void k_convert(const float* __restrict__ src,
                          unsigned short* __restrict__ dst, int n) {
  int i = blockIdx.x * blockDim.x + threadIdx.x;
  if (i < n) dst[i] = f32_bf16(src[i]);
}

// ---------------------------------------------------------------------------
// Pregate GEMM: gates[dir][s][b][n'] = act[b*S+s][:] @ Wih[dir][n'][:]^T + bias
// M = 32768, N = 3072, K = 768.  One wave = 16x64 tile.
// ---------------------------------------------------------------------------
__global__ void k_pregate(const unsigned short* __restrict__ act,
                          const unsigned short* __restrict__ wih,   // [3072][768] bf16
                          const float* __restrict__ bias,           // [3072]
                          float* __restrict__ gates) {
  const int lane = threadIdx.x & 31;
  const int wid  = (blockIdx.x * blockDim.x + threadIdx.x) >> 5;
  const int tn = wid % 48;
  const int tm = wid / 48;
  const int n0 = tn * 64;
  const int m0 = tm * 16;
  const int am = m0 + (lane & 15);
  const int bn = lane & 15;

  v8f acc0 = {}, acc1 = {}, acc2 = {}, acc3 = {};
  for (int k0 = 0; k0 < D_; k0 += 32) {
    __builtin_prefetch((const void*)(act + (size_t)am * D_ + k0 + 64), 0, 1);
    v16bf a  = load_a(act, D_, am, k0, lane);
    v16bf b0 = load_b(wih, D_, n0 +  0 + bn, k0, lane);
    v16bf b1 = load_b(wih, D_, n0 + 16 + bn, k0, lane);
    v16bf b2 = load_b(wih, D_, n0 + 32 + bn, k0, lane);
    v16bf b3 = load_b(wih, D_, n0 + 48 + bn, k0, lane);
    WMMA_BF16(acc0, a, b0);
    WMMA_BF16(acc1, a, b1);
    WMMA_BF16(acc2, a, b2);
    WMMA_BF16(acc3, a, b3);
  }

  const int dir = (n0 >= G4H) ? 1 : 0;
  const int np0 = n0 - dir * G4H;
  const int rhi = (lane >> 4) << 3;
  v8f accs[4] = {acc0, acc1, acc2, acc3};
#pragma unroll
  for (int j = 0; j < 4; ++j) {
    int np = np0 + j * 16 + bn;
    float bv = bias[dir * G4H + np];
#pragma unroll
    for (int r = 0; r < 8; ++r) {
      int m  = m0 + r + rhi;
      int bb = m >> 9;                // m = b*S + s
      int ss = m & (S_ - 1);
      gates[(((size_t)dir * S_ + ss) * B_ + bb) * G4H + np] = accs[j][r] + bv;
    }
  }
}

// ---------------------------------------------------------------------------
// Persistent per-layer scan. Grid = 8 blocks (dir x 4 batch-row tiles),
// 256 threads (8 waves). Per step: WMMA h@Whh^T (A from LDS, B from L2),
// accumulate in place into the async-prefetched pregate LDS tile, LSTM cell
// from LDS, h stays in LDS, c in registers. Next step's pregate tile is
// fetched by the async DMA engine while WMMA runs.
// ---------------------------------------------------------------------------
__global__ void k_scan(const float* __restrict__ gates,        // [2][512][64][1536]
                       const unsigned short* __restrict__ whh, // [2][1536][384] bf16
                       unsigned short* __restrict__ act_out)   // [B*S][768] bf16
{
  extern __shared__ char smem[];
  float*          plds0 = (float*)smem;                        // [16][1536]
  float*          plds1 = (float*)(smem + PLDS_WORDS * 4);     // [16][1536]
  unsigned short* hlds  = (unsigned short*)(smem + 2 * PLDS_WORDS * 4); // [16][392]

  const int tid  = threadIdx.x;
  const int lane = tid & 31;
  const int w    = tid >> 5;                 // wave 0..7
  const int dir  = blockIdx.x >> 2;
  const int m0   = (blockIdx.x & 3) * 16;    // batch-row base
  const unsigned short* W = whh + (size_t)dir * G4H * H_;
  const int colw = w * 192;                  // this wave's 192-column strip
  const int bn   = lane & 15;
  const int rhi  = (lane >> 4) << 3;
  const int am   = lane & 15;                // local A row

  // per-thread cell slots: element e = tid + i*256 of the 16x384 tile
  float creg[24];
  int goff[24], hoff[24], obase[24];
#pragma unroll
  for (int i = 0; i < 24; ++i) {
    int e   = tid + i * 256;
    int row = e / H_;
    int j   = e - row * H_;
    creg[i]  = 0.f;
    goff[i]  = row * G4H + j;
    hoff[i]  = row * HLDS_STRIDE + j;
    obase[i] = ((m0 + row) * S_) * D_ + dir * H_ + j;   // + t_eff*D_ later
  }
  for (int e = tid; e < 16 * HLDS_STRIDE; e += 256) hlds[e] = 0;  // h(-1)=0

  // gates tile for a step is one contiguous 98304B block; async copy is linear
  const unsigned int lbase0 = (unsigned int)(uintptr_t)(void*)plds0;
  const unsigned int lbase1 = (unsigned int)(uintptr_t)(void*)plds1;
  const char* g0 = (const char*)(gates +
      (((size_t)dir * S_ + (dir ? (S_ - 1) : 0)) * B_ + m0) * G4H);
#pragma unroll
  for (int q = 0; q < 24; ++q)          // prologue: prefetch tile for t=0
    async_ld_b128(lbase0 + tid * 16 + q * 4096, g0 + tid * 16 + q * 4096);
  __syncthreads();

  const v8f vzero = {};

  for (int t = 0; t < S_; ++t) {
    // ---- GEMM: acc[12] = h_prev(LDS) @ Whh^T; overlaps in-flight async ----
    v8f acc[12];
#pragma unroll
    for (int j = 0; j < 12; ++j) acc[j] = vzero;

    for (int k0 = 0; k0 < H_; k0 += 32) {
      v16bf a = load_a(hlds, HLDS_STRIDE, am, k0, lane);   // ds_load_b128
#pragma unroll
      for (int j = 0; j < 12; ++j) {
        v16bf bfr = load_b(W, H_, colw + j * 16 + bn, k0, lane);
        WMMA_BF16(acc[j], a, bfr);
      }
    }

    // ---- current pregate tile is fully in LDS after wait+barrier ----
    wait_async0();
    __syncthreads();
    float* pc = (t & 1) ? plds1 : plds0;

    // ---- kick off async fetch of next step's tile into the other buffer ----
    if (t + 1 < S_) {
      const int t_next = dir ? (S_ - 2 - t) : (t + 1);
      unsigned int lb = (t & 1) ? lbase0 : lbase1;
      const char* gn = (const char*)(gates +
          (((size_t)dir * S_ + t_next) * B_ + m0) * G4H);
#pragma unroll
      for (int q = 0; q < 24; ++q)
        async_ld_b128(lb + tid * 16 + q * 4096, gn + tid * 16 + q * 4096);
    }

    // ---- epilogue: gates(LDS) += recurrent WMMA contribution (in place) ----
#pragma unroll
    for (int j = 0; j < 12; ++j) {
      int col = colw + j * 16 + bn;
#pragma unroll
      for (int r = 0; r < 8; ++r) {
        int idx = (rhi + r) * G4H + col;
        pc[idx] += acc[j][r];
      }
    }
    __syncthreads();

    // ---- LSTM cell from LDS; h -> LDS + layer output; c stays in regs ----
    const int t_eff = dir ? (S_ - 1 - t) : t;
#pragma unroll
    for (int i = 0; i < 24; ++i) {
      float gi = pc[goff[i]];
      float gf = pc[goff[i] + H_];
      float gg = pc[goff[i] + 2 * H_];
      float go = pc[goff[i] + 3 * H_];
      float c  = creg[i];
      float si = 1.f / (1.f + __expf(-gi));
      float sf = 1.f / (1.f + __expf(-gf));
      float so = 1.f / (1.f + __expf(-go));
      c = sf * c + si * tanhf(gg);
      float h = so * tanhf(c);
      creg[i] = c;
      unsigned short hb = f32_bf16(h);
      hlds[hoff[i]] = hb;
      act_out[(size_t)obase[i] + (size_t)t_eff * D_] = hb;
    }
    __syncthreads();
  }
}

// ---------------------------------------------------------------------------
// FC: em[row][t] = act[row][:] . fc_w[t][:] + fc_b[t]   (T=9, K=768)
// ---------------------------------------------------------------------------
__global__ void k_fc(const unsigned short* __restrict__ act,
                     const float* __restrict__ fw,
                     const float* __restrict__ fb,
                     float* __restrict__ em) {
  __shared__ float w[T_ * D_];
  for (int i = threadIdx.x; i < T_ * D_; i += blockDim.x) w[i] = fw[i];
  __syncthreads();
  int row = blockIdx.x * blockDim.x + threadIdx.x;
  if (row >= B_ * S_) return;
  const unsigned short* x = act + (size_t)row * D_;
  float acc[T_];
#pragma unroll
  for (int t = 0; t < T_; ++t) acc[t] = fb[t];
  for (int k = 0; k < D_; ++k) {
    float xv = bf16_f32(x[k]);
#pragma unroll
    for (int t = 0; t < T_; ++t) acc[t] += xv * w[t * D_ + k];
  }
#pragma unroll
  for (int t = 0; t < T_; ++t) em[(size_t)row * T_ + t] = acc[t];
}

// ---------------------------------------------------------------------------
// CRF forward (logZ) + gold path score; one wave per batch element.
// ---------------------------------------------------------------------------
__global__ void k_crf(const float* __restrict__ em,
                      const int* __restrict__ y,
                      const float* __restrict__ trans,
                      const float* __restrict__ start_t,
                      const float* __restrict__ end_t,
                      float* __restrict__ perb) {
  const int b = blockIdx.x, lane = threadIdx.x;
  const float* eb = em + (size_t)b * S_ * T_;
  const float NEG = -3.0e38f;
  float tr[T_];
#pragma unroll
  for (int i = 0; i < T_; ++i) tr[i] = (lane < T_) ? trans[i * T_ + lane] : 0.f;

  float alpha = (lane < T_) ? (start_t[lane] + eb[lane]) : NEG;
  for (int t = 1; t < S_; ++t) {
    float v[T_]; float mx = NEG;
#pragma unroll
    for (int i = 0; i < T_; ++i) {
      float ai = __shfl(alpha, i, 32);
      v[i] = ai + tr[i];
      mx = fmaxf(mx, v[i]);
    }
    float ssum = 0.f;
#pragma unroll
    for (int i = 0; i < T_; ++i) ssum += __expf(v[i] - mx);
    float e  = (lane < T_) ? eb[t * T_ + lane] : 0.f;
    float na = mx + __logf(ssum) + e;
    alpha = (lane < T_) ? na : NEG;
  }
  float val = (lane < T_) ? (alpha + end_t[lane]) : NEG;
  float mz = NEG;
#pragma unroll
  for (int i = 0; i < T_; ++i) mz = fmaxf(mz, __shfl(val, i, 32));
  float sz = 0.f;
#pragma unroll
  for (int i = 0; i < T_; ++i) sz += __expf(__shfl(val, i, 32) - mz);
  float logZ = mz + __logf(sz);

  const int* yb = y + b * S_;
  float g = 0.f;
  for (int s = lane; s < S_; s += 32) {
    int ys = yb[s];
    g += eb[s * T_ + ys];
    if (s > 0) g += trans[yb[s - 1] * T_ + ys];
  }
#pragma unroll
  for (int off = 16; off > 0; off >>= 1) g += __shfl_xor(g, off, 32);
  if (lane == 0) {
    g += start_t[yb[0]] + end_t[yb[S_ - 1]];
    perb[b] = logZ - g;
  }
}

__global__ void k_loss(const float* __restrict__ perb, float* __restrict__ out) {
  int lane = threadIdx.x;
  float v = perb[lane] + perb[lane + 32];
#pragma unroll
  for (int off = 16; off > 0; off >>= 1) v += __shfl_xor(v, off, 32);
  if (lane == 0) out[0] = v / (float)B_;
}

// ---------------------------------------------------------------------------
// Viterbi; one wave per batch, byte backpointers, lane-0 backtrace.
// ---------------------------------------------------------------------------
__global__ void k_viterbi(const float* __restrict__ em,
                          const float* __restrict__ trans,
                          const float* __restrict__ start_t,
                          const float* __restrict__ end_t,
                          unsigned char* __restrict__ bp,           // [S][B][T]
                          float* __restrict__ tag_out) {
  const int b = blockIdx.x, lane = threadIdx.x;
  const float* eb = em + (size_t)b * S_ * T_;
  const float NEG = -3.0e38f;
  float tr[T_];
#pragma unroll
  for (int i = 0; i < T_; ++i) tr[i] = (lane < T_) ? trans[i * T_ + lane] : 0.f;

  float v = (lane < T_) ? (start_t[lane] + eb[lane]) : NEG;
  for (int t = 1; t < S_; ++t) {
    float best = NEG; int bi = 0;
#pragma unroll
    for (int i = 0; i < T_; ++i) {
      float cand = __shfl(v, i, 32) + tr[i];
      if (cand > best) { best = cand; bi = i; }    // strict > -> first argmax
    }
    if (lane < T_) {
      bp[((size_t)t * B_ + b) * T_ + lane] = (unsigned char)bi;
      v = best + eb[t * T_ + lane];
    }
  }
  float val = (lane < T_) ? (v + end_t[lane]) : NEG;
  float bm = NEG; int bj = 0;
#pragma unroll
  for (int i = 0; i < T_; ++i) {
    float c = __shfl(val, i, 32);
    if (c > bm) { bm = c; bj = i; }
  }
  if (lane == 0) {
    int tag = bj;
    for (int t = S_ - 1; t >= 1; --t) {
      tag_out[b * S_ + t] = (float)tag;
      tag = bp[((size_t)t * B_ + b) * T_ + tag];
    }
    tag_out[b * S_] = (float)tag;
  }
}

// ---------------------------------------------------------------------------
// Host orchestration: 11 launches total.
// ---------------------------------------------------------------------------
extern "C" void kernel_launch(void* const* d_in, const int* in_sizes, int n_in,
                              void* d_out, int out_size, void* d_ws, size_t ws_size,
                              hipStream_t stream) {
  const float* bert  = (const float*)d_in[0];
  const int*   y     = (const int*)  d_in[1];
  const float* Wih   = (const float*)d_in[2];
  const float* Whh   = (const float*)d_in[3];
  const float* lb    = (const float*)d_in[4];
  const float* fw    = (const float*)d_in[5];
  const float* fb    = (const float*)d_in[6];
  const float* trans = (const float*)d_in[7];
  const float* st    = (const float*)d_in[8];
  const float* et    = (const float*)d_in[9];
  float* out = (float*)d_out;
  (void)in_sizes; (void)n_in; (void)out_size; (void)ws_size;

  char* ws = (char*)d_ws;
  size_t off = 0;
  auto carve = [&](size_t bytes) -> char* {
    off = (off + 255) & ~(size_t)255;
    char* p = ws + off;
    off += bytes;
    return p;
  };
  float*          gates = (float*)         carve((size_t)2 * S_ * B_ * G4H * 4); // 403 MB
  unsigned short* actA  = (unsigned short*)carve((size_t)B_ * S_ * D_ * 2);      //  50 MB
  unsigned short* actB  = (unsigned short*)carve((size_t)B_ * S_ * D_ * 2);      //  50 MB
  unsigned short* wihb  = (unsigned short*)carve((size_t)L_ * 2 * G4H * D_ * 2);
  unsigned short* whhb  = (unsigned short*)carve((size_t)L_ * 2 * G4H * H_ * 2);
  float*          em    = (float*)         carve((size_t)B_ * S_ * T_ * 4);
  unsigned char*  bp    = (unsigned char*) carve((size_t)S_ * B_ * T_);
  float*          perb  = (float*)         carve((size_t)B_ * 4);

  // allow >64KB dynamic LDS for the scan kernel (stateless, capture-safe)
  hipFuncSetAttribute(reinterpret_cast<const void*>(k_scan),
                      hipFuncAttributeMaxDynamicSharedMemorySize, SMEM_BYTES);

  // bf16 conversions of activations and LSTM weights
  {
    int n = B_ * S_ * D_;
    k_convert<<<(n + 255) / 256, 256, 0, stream>>>(bert, actA, n);
    n = L_ * 2 * G4H * D_;
    k_convert<<<(n + 255) / 256, 256, 0, stream>>>(Wih, wihb, n);
    n = L_ * 2 * G4H * H_;
    k_convert<<<(n + 255) / 256, 256, 0, stream>>>(Whh, whhb, n);
  }

  for (int l = 0; l < L_; ++l) {
    const unsigned short* in_act  = l ? actB : actA;
    unsigned short*       out_act = l ? actA : actB;
    const unsigned short* wl = wihb + (size_t)l * 2 * G4H * D_;
    const unsigned short* rl = whhb + (size_t)l * 2 * G4H * H_;
    const float*          bl = lb   + (size_t)l * 2 * G4H;

    // hoisted input projection for all timesteps, both directions (+bias)
    k_pregate<<<12288, 256, 0, stream>>>(in_act, wl, bl, gates);

    // whole bidirectional scan of this layer in ONE launch
    k_scan<<<8, 256, SMEM_BYTES, stream>>>(gates, rl, out_act);
  }

  k_fc<<<(B_ * S_ + 255) / 256, 256, 0, stream>>>(actA, fw, fb, em);
  k_crf<<<B_, 32, 0, stream>>>(em, y, trans, st, et, perb);
  k_loss<<<1, 32, 0, stream>>>(perb, out);
  k_viterbi<<<B_, 32, 0, stream>>>(em, trans, st, et, bp, out + 1);
}
